// GptOssMLAAttention_46059229283041
// MI455X (gfx1250) — compile-verified, matrix operand order checked by reading, not run on hardware
//
#include <hip/hip_runtime.h>
#include <hip/hip_bf16.h>
#include <stdint.h>

// ---------------------------------------------------------------------------
// MLA attention forward for MI455X (gfx1250, wave32, WMMA + TDM).
// All matrix math via v_wmma_f32_16x16x32_f16; LDS tiles stored in WMMA
// fragment order so operand loads are b128 vector loads, not scalar gathers.
// ---------------------------------------------------------------------------

typedef __attribute__((ext_vector_type(16))) _Float16 v16h;
typedef __attribute__((ext_vector_type(8)))  _Float16 v8h;
typedef __attribute__((ext_vector_type(8)))  float    v8f;
typedef __attribute__((ext_vector_type(4)))  unsigned u32x4;
typedef __attribute__((ext_vector_type(8)))  int      i32x8;
typedef __attribute__((ext_vector_type(4)))  int      i32x4;

#define B_    2
#define S_    1024
#define HID_  4096
#define H_    32
#define NOPE_ 128
#define ROPE_ 64
#define VD_   128
#define R_    512
#define QKD_  192            // NOPE + ROPE
#define DQK_  576            // R + ROPE
#define SCALE_ 0.07216878364870323f   // 192^-0.5

#if defined(__has_builtin)
#if __has_builtin(__builtin_amdgcn_tensor_load_to_lds) && __has_builtin(__builtin_amdgcn_s_wait_tensorcnt)
#define HAS_TDM 1
#endif
#endif
#ifndef HAS_TDM
#define HAS_TDM 0
#endif

// ---------------------------------------------------------------------------
// WMMA helpers. Fragment order: a 16x16x32 operand fragment is 16 halves per
// lane; element j of lane L maps to (per ISA 7.12.2):
//   row/col = L%16 (+ tile), K = frag_k(j, L/16)
// We invert that mapping at staging time so loads are contiguous per lane.
// ---------------------------------------------------------------------------
__device__ __forceinline__ v8f wmma_f16(v16h a, v16h b, v8f c) {
    return __builtin_amdgcn_wmma_f32_16x16x32_f16(
        false, a, false, b, (short)0, c, false, false);
}

// fragment slot j (0..15) holding K index k (0..31)
__device__ __forceinline__ int frag_j(int k) {
    return 2 * (((k & 7) >> 1) + ((k >> 4) << 2)) + (k & 1);
}
// lane offset (0 or 16) for K index k
__device__ __forceinline__ int frag_lh(int k) { return ((k >> 3) & 1) << 4; }

__device__ __forceinline__ void store_pair_f(_Float16* base, float lo, float hi) {
    union { _Float16 h[2]; unsigned u; } p;
    p.h[0] = (_Float16)lo; p.h[1] = (_Float16)hi;
    *reinterpret_cast<unsigned*>(base) = p.u;
}
__device__ __forceinline__ void store_pair_h(_Float16* base, _Float16 lo, _Float16 hi) {
    union { _Float16 h[2]; unsigned u; } p;
    p.h[0] = lo; p.h[1] = hi;
    *reinterpret_cast<unsigned*>(base) = p.u;
}

// ---------------------------------------------------------------------------
// Tensor Data Mover: 2-D tile (rows x dim0 halves) global -> LDS, per ISA 8.3/8.4.
// 6-arg builtin form (clang-23 / therock-10.0 lane).
// ---------------------------------------------------------------------------
#if HAS_TDM
__device__ __forceinline__ void tdm_load_2d(unsigned lds_off, const void* gptr,
                                            unsigned dim0, unsigned dim1,
                                            unsigned stride0,
                                            unsigned tile0, unsigned tile1)
{
    unsigned long long ga = (unsigned long long)(uintptr_t)gptr;
    u32x4 g0;
    g0[0] = 1u;                                        // count=1, user mode
    g0[1] = lds_off;                                   // lds_addr
    g0[2] = (unsigned)(ga & 0xffffffffu);              // global_addr lo
    g0[3] = (unsigned)((ga >> 32) & 0x01ffffffu) | 0x80000000u;  // addr hi | type=2
    i32x8 g1 = {};
    g1[0] = (int)(1u << 16);                           // data_size = 2 bytes
    g1[1] = (int)(dim0 << 16);                         // tensor_dim0[15:0]
    g1[2] = (int)((dim0 >> 16) | (dim1 << 16));        // dim0 hi | tensor_dim1 lo
    g1[3] = (int)((dim1 >> 16) | (tile0 << 16));       // dim1 hi | tile_dim0
    g1[4] = (int)(tile1 & 0xffffu);                    // tile_dim1 (tile_dim2=0)
    g1[5] = (int)stride0;                              // tensor_dim0_stride lo
    g1[6] = 0; g1[7] = 0;
    i32x4 z4 = {};
    i32x8 z8 = {};
    __builtin_amdgcn_tensor_load_to_lds(g0, g1, z4, z4, z8, 0);
}
#endif

// ---------------------------------------------------------------------------
// Generic tiled GEMM: C[M,N] = alpha * A[M,K] * B[K,N]; f32 in, f32/f16 out.
// 64x64 block tile, k-step 32, 8 waves. Tiles staged in fragment order.
// ---------------------------------------------------------------------------
#define GM 64
#define GN 64
#define GK 32

__global__ __launch_bounds__(256)
void gemm_wmma(const float* __restrict__ A, int lda, long long strideA,
               const float* __restrict__ Bm, int ldb, long long strideB,
               void* __restrict__ C, int ldc, long long strideC, int c_f16,
               float alpha, int M, int N, int K)
{
    __shared__ __align__(32) _Float16 As[4 * 32 * 16];   // 4 m-tiles, frag order
    __shared__ __align__(32) _Float16 Bs[4 * 32 * 16];   // 4 n-tiles, frag order

    int bz = blockIdx.z;
    A  += (size_t)bz * strideA;
    Bm += (size_t)bz * strideB;

    int tid = threadIdx.x, wid = tid >> 5, lane = tid & 31;
    int mt = wid & 3, nt = wid >> 2;
    int row0 = blockIdx.y * GM, col0 = blockIdx.x * GN;

    int ar = (tid * 8) >> 5;        // A: row 0..63
    int ac = (tid * 8) & 31;        // A: col 0,8,16,24
    int br = tid >> 3;              // B: row 0..31
    int bc = (tid & 7) * 8;         // B: col 0..56

    v8f acc0 = {}, acc1 = {};

    for (int k0 = 0; k0 < K; k0 += GK) {
        // ---- stage A 64x32 -> fragment order (b128 loads, paired b32 stores)
        {
            const float* src = &A[(size_t)(row0 + ar) * lda + (k0 + ac)];
            float4 f0 = *reinterpret_cast<const float4*>(src);
            float4 f1 = *reinterpret_cast<const float4*>(src + 4);
            float fv[8] = {f0.x, f0.y, f0.z, f0.w, f1.x, f1.y, f1.z, f1.w};
            int tile = ar >> 4, m = ar & 15;
#pragma unroll
            for (int p = 0; p < 4; ++p) {
                int k = ac + 2 * p;
                int idx = ((tile << 5) + m + frag_lh(k)) * 16 + frag_j(k);
                store_pair_f(&As[idx], fv[2 * p], fv[2 * p + 1]);
            }
        }
        // ---- stage B 32x64 -> fragment order
        {
            const float* src = &Bm[(size_t)(k0 + br) * ldb + (col0 + bc)];
            float4 f0 = *reinterpret_cast<const float4*>(src);
            float4 f1 = *reinterpret_cast<const float4*>(src + 4);
            float fv[8] = {f0.x, f0.y, f0.z, f0.w, f1.x, f1.y, f1.z, f1.w};
            int jj = frag_j(br), lh = frag_lh(br);
#pragma unroll
            for (int e = 0; e < 8; ++e) {
                int n = bc + e;
                Bs[(((n >> 4) << 5) + (n & 15) + lh) * 16 + jj] = (_Float16)fv[e];
            }
        }
        if (k0 + GK < K) {
            __builtin_prefetch(&A[(size_t)(row0 + ar) * lda + (k0 + GK + ac)], 0, 0);
            __builtin_prefetch(&Bm[(size_t)(k0 + GK + br) * ldb + (col0 + bc)], 0, 0);
        }
        __syncthreads();

        v16h af  = *reinterpret_cast<const v16h*>(&As[((mt << 5) + lane) << 4]);
        v16h bf0 = *reinterpret_cast<const v16h*>(&Bs[(((nt * 2 + 0) << 5) + lane) << 4]);
        v16h bf1 = *reinterpret_cast<const v16h*>(&Bs[(((nt * 2 + 1) << 5) + lane) << 4]);
        acc0 = wmma_f16(af, bf0, acc0);
        acc1 = wmma_f16(af, bf1, acc1);
        __syncthreads();
    }

    int nA = col0 + nt * 32 + (lane & 15);
    int mB = row0 + mt * 16 + ((lane >> 4) << 3);
#pragma unroll
    for (int v = 0; v < 8; ++v) {
        int m = mB + v;
        float x0 = acc0[v] * alpha;
        float x1 = acc1[v] * alpha;
        if (c_f16) {
            _Float16* C16 = (_Float16*)C + (size_t)bz * strideC;
            C16[(size_t)m * ldc + nA]      = (_Float16)x0;
            C16[(size_t)m * ldc + nA + 16] = (_Float16)x1;
        } else {
            float* C32 = (float*)C + (size_t)bz * strideC;
            C32[(size_t)m * ldc + nA]      = x0;
            C32[(size_t)m * ldc + nA + 16] = x1;
        }
    }
}

// ---------------------------------------------------------------------------
// RoPE + assembly: builds f16 q_full rope part (pre-scaled) and f16 k_full.
// ---------------------------------------------------------------------------
__global__ __launch_bounds__(256)
void rope_assemble(const float* __restrict__ q_raw,   // [B*S, H*QKD]
                   const float* __restrict__ lat,     // [B*S, DQK]
                   const int*   __restrict__ pos,     // [S]
                   _Float16* __restrict__ qf,         // [B*S, H, DQK]
                   _Float16* __restrict__ kf)         // [B*S, DQK]
{
    __shared__ float cs[32], sn[32];
    int token = blockIdx.x;
    int s     = token & (S_ - 1);
    int tid   = threadIdx.x;

    if (tid < 32) {
        float inv = __powf(10000.0f, -((float)tid) / 32.0f);
        float ang = (float)pos[s] * inv;
        cs[tid] = __cosf(ang);
        sn[tid] = __sinf(ang);
    }
    __syncthreads();

    const float* qrow = q_raw + (size_t)token * (H_ * QKD_);
    const float* lrow = lat   + (size_t)token * DQK_;
    _Float16* qo = qf + (size_t)token * (H_ * DQK_);
    _Float16* ko = kf + (size_t)token * DQK_;

    for (int i = tid; i < H_ * 32; i += 256) {
        int h = i >> 5, j = i & 31;
        float x1 = qrow[h * QKD_ + NOPE_ + j];
        float x2 = qrow[h * QKD_ + NOPE_ + 32 + j];
        qo[h * DQK_ + R_ + j]      = (_Float16)((x1 * cs[j] - x2 * sn[j]) * SCALE_);
        qo[h * DQK_ + R_ + 32 + j] = (_Float16)((x2 * cs[j] + x1 * sn[j]) * SCALE_);
    }
    if (tid < 32) {
        int j = tid;
        float x1 = lrow[R_ + j];
        float x2 = lrow[R_ + 32 + j];
        ko[R_ + j]      = (_Float16)(x1 * cs[j] - x2 * sn[j]);
        ko[R_ + 32 + j] = (_Float16)(x2 * cs[j] + x1 * sn[j]);
    }
    for (int i = tid; i < R_; i += 256)
        ko[i] = (_Float16)lrow[i];
}

// ---------------------------------------------------------------------------
// Flash attention: 32-query tile/block, 32-key steps, causal + sink.
// Q staged once (fragment order); K tile landed via TDM into plain LDS then
// re-swizzled into K^T (scores) and V (P*V) fragment views.
// ---------------------------------------------------------------------------
#define AQ 32
#define AK 32
#define KSTEPS (DQK_ / 32)   // 18

__global__ __launch_bounds__(256)
void attn_wmma(const _Float16* __restrict__ qf,   // [B*S, H, DQK] (pre-scaled)
               const _Float16* __restrict__ kf,   // [B*S, DQK]
               const float*    __restrict__ sinks,// [H]
               float* __restrict__ attn)          // [B*S, H, R]
{
    __shared__ __align__(32) _Float16 Pl[AK * DQK_];             // plain landing tile
    __shared__ __align__(32) _Float16 Qz[2 * KSTEPS * 32 * 16];  // Q frags
    __shared__ __align__(32) _Float16 Kz[KSTEPS * 2 * 32 * 16];  // K^T frags (scores B)
    __shared__ __align__(32) _Float16 Vz[32 * 32 * 16];          // V frags (PV B)
    __shared__ __align__(32) _Float16 Pz[2 * 32 * 16];           // P frags (PV A)
    __shared__ float Ss[AQ * AK];
    __shared__ float rowm[AQ], rowl[AQ], rowsc[AQ];

    int qt = blockIdx.x, h = blockIdx.y, b = blockIdx.z;
    int q0 = qt * AQ;
    int tid = threadIdx.x, wid = tid >> 5, lane = tid & 31;
    int mt = wid & 1, ng = wid >> 1;
    int rowA = ((lane >> 4) << 3);

    if (tid < AQ) { rowm[tid] = sinks[h]; rowl[tid] = 1.0f; }

    // Pl pass coords: 8 threads/row, 9 chunks of 8 halves each
    int pr = tid >> 3;
    int pc = (tid & 7) * 72;

    const size_t qrow0 = ((size_t)(b * S_ + q0) * H_ + h) * DQK_;

    // ---- land Q tile (strided rows) into Pl ----
#if HAS_TDM
    if (wid == 0) {
        tdm_load_2d((unsigned)(uintptr_t)Pl, qf + qrow0,
                    DQK_, AQ, (unsigned)(H_ * DQK_), DQK_, AQ);
        __builtin_amdgcn_s_wait_tensorcnt(0);
    }
#else
    for (int qq = 0; qq < 9; ++qq) {
        int d0 = pc + qq * 8;
        *reinterpret_cast<v8h*>(&Pl[pr * DQK_ + d0]) =
            *reinterpret_cast<const v8h*>(&qf[qrow0 + (size_t)pr * (H_ * DQK_) + d0]);
    }
#endif
    __syncthreads();

    // ---- swizzle Pl -> Qz ----
    {
        int tile = pr >> 4, m = pr & 15;
        for (int qq = 0; qq < 9; ++qq) {
            int d0 = pc + qq * 8;
            v8h v = *reinterpret_cast<const v8h*>(&Pl[pr * DQK_ + d0]);
#pragma unroll
            for (int p = 0; p < 4; ++p) {
                int d = d0 + 2 * p;
                int ks = d >> 5, k = d & 31;
                int idx = (((tile * KSTEPS + ks) << 5) + m + frag_lh(k)) * 16 + frag_j(k);
                store_pair_h(&Qz[idx], v[2 * p], v[2 * p + 1]);
            }
        }
    }
    __syncthreads();

    v8f acc[8];
#pragma unroll
    for (int t = 0; t < 8; ++t) acc[t] = v8f{};

    for (int k0 = 0; k0 < q0 + AQ; k0 += AK) {
        __syncthreads();   // protect Pl/Kz/Vz/Pz reuse across iterations

        // ---- land K tile into Pl ----
#if HAS_TDM
        if (wid == 0) {
            tdm_load_2d((unsigned)(uintptr_t)Pl,
                        kf + ((size_t)(b * S_ + k0)) * DQK_,
                        DQK_, AK, DQK_, DQK_, AK);
            __builtin_amdgcn_s_wait_tensorcnt(0);
        }
#else
        for (int qq = 0; qq < 9; ++qq) {
            int d0 = pc + qq * 8;
            *reinterpret_cast<v8h*>(&Pl[pr * DQK_ + d0]) =
                *reinterpret_cast<const v8h*>(&kf[((size_t)(b * S_ + k0 + pr)) * DQK_ + d0]);
        }
#endif
        __syncthreads();

        // ---- swizzle Pl -> Kz (k=dim, n=key) and Vz (k=key, n=dim<512) ----
        {
            int key = pr, lhk = frag_lh(key), jk = frag_j(key);
            for (int qq = 0; qq < 9; ++qq) {
                int d0 = pc + qq * 8;
                v8h v = *reinterpret_cast<const v8h*>(&Pl[key * DQK_ + d0]);
#pragma unroll
                for (int p = 0; p < 4; ++p) {
                    int d = d0 + 2 * p;
                    int ks = d >> 5, k = d & 31;
                    int idx = (((ks * 2 + (key >> 4)) << 5) + (key & 15) + frag_lh(k)) * 16 + frag_j(k);
                    store_pair_h(&Kz[idx], v[2 * p], v[2 * p + 1]);
                }
                if (d0 < R_) {
#pragma unroll
                    for (int e = 0; e < 8; ++e) {
                        int n = d0 + e;
                        Vz[(((n >> 4) << 5) + (n & 15) + lhk) * 16 + jk] = v[e];
                    }
                }
            }
        }
        __syncthreads();

        // ---- scores (waves 0-3, one 16x16 tile each) ----
        if (wid < 4) {
            int smt = wid & 1, snt = (wid >> 1) & 1;
            v8f sc = {};
#pragma unroll
            for (int ks = 0; ks < KSTEPS; ++ks) {
                v16h af = *reinterpret_cast<const v16h*>(&Qz[(((smt * KSTEPS + ks) << 5) + lane) << 4]);
                v16h bf = *reinterpret_cast<const v16h*>(&Kz[(((ks * 2 + snt) << 5) + lane) << 4]);
                sc = wmma_f16(af, bf, sc);
            }
#pragma unroll
            for (int v = 0; v < 8; ++v) {
                int m = smt * 16 + rowA + v;
                int n = snt * 16 + (lane & 15);
                float x = sc[v];
                if (k0 + n > q0 + m) x = -3.0e38f;   // causal mask
                Ss[m * AK + n] = x;
            }
        }
        __syncthreads();

        // ---- online softmax (one row per thread), P written in frag order ----
        if (tid < AQ) {
            int m = tid;
            float mx = rowm[m];
            for (int n = 0; n < AK; ++n) mx = fmaxf(mx, Ss[m * AK + n]);
            float scale = __expf(rowm[m] - mx);
            float l = rowl[m] * scale;
            int tile = m >> 4, ml = m & 15;
            for (int n = 0; n < AK; n += 2) {
                float p0 = __expf(Ss[m * AK + n] - mx);
                float p1 = __expf(Ss[m * AK + n + 1] - mx);
                l += p0 + p1;
                int idx = ((tile << 5) + ml + frag_lh(n)) * 16 + frag_j(n);
                store_pair_f(&Pz[idx], p0, p1);
            }
            rowm[m] = mx; rowl[m] = l; rowsc[m] = scale;
        }
        __syncthreads();

        // ---- rescale accumulators, then P x V ----
#pragma unroll
        for (int t = 0; t < 8; ++t)
#pragma unroll
            for (int v = 0; v < 8; ++v)
                acc[t][v] *= rowsc[mt * 16 + rowA + v];

        {
            v16h pa = *reinterpret_cast<const v16h*>(&Pz[((mt << 5) + lane) << 4]);
#pragma unroll
            for (int t = 0; t < 8; ++t) {
                v16h vb = *reinterpret_cast<const v16h*>(&Vz[((((ng << 3) + t) << 5) + lane) << 4]);
                acc[t] = wmma_f16(pa, vb, acc[t]);
            }
        }
    }

    // ---- epilogue ----
#pragma unroll
    for (int t = 0; t < 8; ++t) {
        int n = (ng * 8 + t) * 16 + (lane & 15);
#pragma unroll
        for (int v = 0; v < 8; ++v) {
            int m = mt * 16 + rowA + v;
            float o = acc[t][v] / rowl[m];
            attn[((size_t)(b * S_ + q0 + m) * H_ + h) * R_ + n] = o;
        }
    }
}

// ---------------------------------------------------------------------------
// Host launcher
// ---------------------------------------------------------------------------
extern "C" void kernel_launch(void* const* d_in, const int* in_sizes, int n_in,
                              void* d_out, int out_size, void* d_ws, size_t ws_size,
                              hipStream_t stream)
{
    const float* hidden = (const float*)d_in[0];
    const int*   pos    = (const int*)  d_in[1];
    const float* Wq     = (const float*)d_in[2];
    const float* Wkva   = (const float*)d_in[3];
    const float* w_kc   = (const float*)d_in[4];
    const float* w_vc   = (const float*)d_in[5];
    const float* sinks  = (const float*)d_in[6];
    const float* Wo     = (const float*)d_in[7];
    float* out = (float*)d_out;

    const int T = B_ * S_;

    char* ws = (char*)d_ws;
    size_t off = 0;
    auto carve = [&](size_t bytes) -> void* {
        void* p = ws + off;
        off = (off + bytes + 255) & ~(size_t)255;
        return p;
    };
    float*    q_raw  = (float*)   carve((size_t)T * H_ * QKD_ * 4);
    float*    lat    = (float*)   carve((size_t)T * DQK_ * 4);
    _Float16* q_full = (_Float16*)carve((size_t)T * H_ * DQK_ * 2);
    _Float16* k_full = (_Float16*)carve((size_t)T * DQK_ * 2);
    float*    attn   = (float*)   carve((size_t)T * H_ * R_ * 4);
    float*    ov     = (float*)   carve((size_t)T * H_ * VD_ * 4);

    dim3 blk(256);

    gemm_wmma<<<dim3((H_ * QKD_) / GN, T / GM, 1), blk, 0, stream>>>(
        hidden, HID_, 0, Wq, H_ * QKD_, 0,
        q_raw, H_ * QKD_, 0, 0, 1.0f, T, H_ * QKD_, HID_);

    gemm_wmma<<<dim3(DQK_ / GN, T / GM, 1), blk, 0, stream>>>(
        hidden, HID_, 0, Wkva, DQK_, 0,
        lat, DQK_, 0, 0, 1.0f, T, DQK_, HID_);

    rope_assemble<<<dim3(T), blk, 0, stream>>>(q_raw, lat, pos, q_full, k_full);

    gemm_wmma<<<dim3(R_ / GN, T / GM, H_), blk, 0, stream>>>(
        q_raw, H_ * QKD_, QKD_,
        w_kc, R_, (long long)NOPE_ * R_,
        q_full, H_ * DQK_, DQK_, 1,
        SCALE_, T, R_, NOPE_);

    attn_wmma<<<dim3(S_ / AQ, H_, B_), blk, 0, stream>>>(q_full, k_full, sinks, attn);

    gemm_wmma<<<dim3(VD_ / GN, T / GM, H_), blk, 0, stream>>>(
        attn, H_ * R_, R_,
        w_vc, VD_, (long long)R_ * VD_,
        ov, H_ * VD_, VD_, 0,
        1.0f, T, VD_, R_);

    gemm_wmma<<<dim3(HID_ / GN, T / GM, 1), blk, 0, stream>>>(
        ov, H_ * VD_, 0, Wo, HID_, 0,
        out, HID_, 0, 0, 1.0f, T, HID_, H_ * VD_);
}
